// CLF_K_QP_Net_12025908429478
// MI455X (gfx1250) — matrix operand order repaired and checked by reading, not compiled, not run on hardware
//
#include <hip/hip_runtime.h>

typedef _Float16 half_t;
typedef __attribute__((ext_vector_type(16))) _Float16 v16h;
typedef __attribute__((ext_vector_type(8)))  _Float16 v8h;
typedef __attribute__((ext_vector_type(8)))  float    v8f;
typedef __attribute__((ext_vector_type(4)))  float    v4f;

#define BDIM   128   // 4 waves of 32
#define WAVES  4
#define SPB    64    // samples per block (4 waves x 16 rows)

__device__ __forceinline__ v8f wmma_f16(v16h a, v16h b, v8f c) {
  return __builtin_amdgcn_wmma_f32_16x16x32_f16(false, a, false, b, (short)0, c, false, false);
}

// Branch-free tanh: gfx1250 V_TANH_F32 (confirmed in codegen), else exp2/rcp.
__device__ __forceinline__ float fast_tanh(float x) {
#if __has_builtin(__builtin_amdgcn_tanhf)
  return __builtin_amdgcn_tanhf(x);
#else
  float e = __builtin_amdgcn_exp2f(x * 2.885390081777927f);   // exp(2x)
  return 1.f - 2.f * __builtin_amdgcn_rcpf(e + 1.f);
#endif
}

__device__ __forceinline__ v16h cat16(v8h lo, v8h hi) {
  return __builtin_shufflevector(lo, hi, 0,1,2,3,4,5,6,7,8,9,10,11,12,13,14,15);
}

// A fragment (16x32 f16, K-chunk kt) from 16x128 row-major f16 LDS buffer.
// a[0..7] = buf[base..base+7], a[8..15] = buf[base+16..base+23]; base 16B aligned.
__device__ __forceinline__ v16h afrag(const half_t* buf, int kt, int hl, int n16) {
  const v8h* p = (const v8h*)(buf + n16 * 128 + 32 * kt + 8 * hl);
  return cat16(p[0], p[2]);
}
// Same but applies s = t - t^3 elementwise (packed f16 math).
__device__ __forceinline__ v16h afrag_s(const half_t* buf, int kt, int hl, int n16) {
  const v8h* p = (const v8h*)(buf + n16 * 128 + 32 * kt + 8 * hl);
  v8h lo = p[0], hi = p[2];
  lo = lo - lo * lo * lo;
  hi = hi - hi * hi * hi;
  return cat16(lo, hi);
}
// B fragment: b[0..7] = base[0..7], b[8..15] = base[16..23] (base = row + 32*kt + 8*hl).
__device__ __forceinline__ v16h bfrag(const half_t* base) {
  const v8h* p = (const v8h*)base;
  return cat16(p[0], p[2]);
}

// tanh(X @ W^T + b): W stored zero-padded [128][32] f16 in LDS with the bias
// folded into column 16 (matched by xa[8] == 1.0 on hl==0 lanes). C starts at
// inline 0 -> no bias splat movs.
__device__ __forceinline__ void hidden16(const v16h& xa, const half_t* sW32,
                                         half_t* buf, int hl, int n16) {
#pragma unroll
  for (int nt = 0; nt < 8; ++nt) {
    int hcol = 16 * nt + n16;
    v8f acc;
#pragma unroll
    for (int v = 0; v < 8; ++v) acc[v] = 0.f;
    v16h bf = bfrag(sW32 + hcol * 32 + 8 * hl);
    acc = wmma_f16(xa, bf, acc);
#pragma unroll
    for (int v = 0; v < 8; ++v) {
      buf[(v + 8 * hl) * 128 + hcol] = (half_t)fast_tanh(acc[v]);
    }
  }
}

__global__ __launch_bounds__(BDIM) void clf_k_qp_kernel(
    const float* __restrict__ x,
    const float* __restrict__ V1_w,  const float* __restrict__ V1_b,
    const float* __restrict__ V2_w,  const float* __restrict__ V2_b,
    const float* __restrict__ K1a_w, const float* __restrict__ K1a_b,
    const float* __restrict__ K1b_w, const float* __restrict__ K1b_b,
    const float* __restrict__ K2a_w, const float* __restrict__ K2a_b,
    const float* __restrict__ K2b_w, const float* __restrict__ K2b_b,
    const float* __restrict__ A_dyn, const float* __restrict__ B_dyn,
    const float* __restrict__ x_goal, const float* __restrict__ u_eq,
    float* __restrict__ out, int B)
{
  extern __shared__ char smem_raw[];
  half_t* sV2w = (half_t*)smem_raw;          // 128x128  [h'][h]   (B for GEMM2)
  half_t* sV2T = sV2w + 128 * 128;           // 128x128  [k][h]    (B for GEMM3)
  half_t* sV1w = sV2T + 128 * 128;           // 128x32   [h][k], col16=bias, rest pad 0
  half_t* sV1T = sV1w + 128 * 32;            // 16x128   [n][k]    (B for GEMM4)
  half_t* sK1a = sV1T + 16 * 128;            // 128x32   col16=bias
  half_t* sK2a = sK1a + 128 * 32;            // 128x32   col16=bias
  half_t* sK1b = sK2a + 128 * 32;            // 16x128   [n][h]
  half_t* sK2b = sK1b + 16 * 128;            // 16x128
  half_t* sAd  = sK2b + 16 * 128;            // 16x32    [n][k] zero-padded (col16 = 0!)

  const int tid  = threadIdx.x;
  const int wave = tid >> 5;
  const int lane = tid & 31;
  const int hl   = lane >> 4;
  const int n16  = lane & 15;

  // per-wave staging (16B-aligned): bufA(t1), bufB(t2/s/p/hidden), 4 f32 tiles
  half_t* pwbase = sAd + 16 * 32;
  half_t* bufA = pwbase + wave * 6144;       // 2048 halves
  half_t* bufB = bufA + 2048;                // 2048 halves
  float*  sGV  = (float*)(bufB + 2048);      // 4 x 256 f32
  float*  sK1t = sGV + 256;
  float*  sK2t = sK1t + 256;
  float*  sF0  = sK2t + 256;

  float* ftab  = (float*)(pwbase + WAVES * 6144);
  float* sV2b  = ftab;                       // 128
  float* sK1bb = sV2b + 128;                 // 16
  float* sK2bb = sK1bb + 16;                 // 16
  float* sBd   = sK2bb + 16;                 // 16x2
  float* sXg   = sBd + 32;                   // 16
  float* sUe   = sXg + 16;                   // 2

  // ---- cooperative weight staging (fp32 -> f16, transposed/padded/bias-folded) ----
  for (int i = tid; i < 128 * 128; i += BDIM) {
    sV2w[i] = (half_t)V2_w[i];
    sV2T[i] = (half_t)V2_w[(i & 127) * 128 + (i >> 7)];   // [k][h] = V2_w[h][k]
  }
  for (int i = tid; i < 128 * 32; i += BDIM) {            // K=16 pad to 32, bias in col 16
    int h = i >> 5, k = i & 31;
    int src = h * 16 + (k & 15);                          // always-valid address
    float w1 = V1_w[src],  b1 = V1_b[h];
    float wa = K1a_w[src], ba = K1a_b[h];
    float wc = K2a_w[src], bc = K2a_b[h];
    float s1 = (k < 16) ? w1 : ((k == 16) ? b1 : 0.f);
    float sa = (k < 16) ? wa : ((k == 16) ? ba : 0.f);
    float sc = (k < 16) ? wc : ((k == 16) ? bc : 0.f);
    sV1w[i] = (half_t)s1; sK1a[i] = (half_t)sa; sK2a[i] = (half_t)sc;
  }
  for (int i = tid; i < 2048; i += BDIM) {
    sV1T[i] = (half_t)V1_w[(i & 127) * 16 + (i >> 7)];    // [n][k] = V1_w[k][n]
    sK1b[i] = (half_t)K1b_w[i]; sK2b[i] = (half_t)K2b_w[i];
  }
  for (int i = tid; i < 16 * 32; i += BDIM) {
    int n = i >> 5, k = i & 31;
    sAd[i] = (k < 16) ? (half_t)A_dyn[n * 16 + (k & 15)] : (half_t)0.f;
  }
  for (int i = tid; i < 128; i += BDIM) sV2b[i] = V2_b[i];
  if (tid < 16) { sK1bb[tid] = K1b_b[tid]; sK2bb[tid] = K2b_b[tid]; sXg[tid] = x_goal[tid]; }
  if (tid < 32) sBd[tid] = B_dyn[tid];
  if (tid < 2)  sUe[tid] = u_eq[tid];
  __syncthreads();

  const int tileBase = blockIdx.x * SPB + wave * 16;
  if (tileBase >= B) return;

  // ---- X fragment (16x16 padded to 16x32), a[8] = 1.0 on hl==0 (bias row K=16) ----
  v16h xa;
  {
    const float* xr = x + (size_t)(tileBase + n16) * 16;
    const v4f* xp = (const v4f*)(xr + 8 * hl);
    v4f x0 = xp[0], x1 = xp[1];
#pragma unroll
    for (int j = 0; j < 4; ++j) {
      xa[j]     = (half_t)x0[j];
      xa[4 + j] = (half_t)x1[j];
      xa[8 + j]  = (half_t)0.f;
      xa[12 + j] = (half_t)0.f;
    }
    xa[8] = (hl == 0) ? (half_t)1.f : (half_t)0.f;   // K=16 -> multiplies bias column
  }

  // ---- GEMM1: t1 = tanh(x @ V1_w^T + b) -> bufA (8 wmma, bias folded) ----
  hidden16(xa, sV1w, bufA, hl, n16);

  // ---- GEMM2: t2 = tanh(t1 @ V2_w^T + b) -> bufB (32 wmma) ----
  {
    v16h a1[4];
#pragma unroll
    for (int kt = 0; kt < 4; ++kt) a1[kt] = afrag(bufA, kt, hl, n16);
#pragma unroll
    for (int nt = 0; nt < 8; ++nt) {
      int hc = 16 * nt + n16;
      float bias = sV2b[hc];
      v8f acc;
#pragma unroll
      for (int v = 0; v < 8; ++v) acc[v] = bias;
#pragma unroll
      for (int kt = 0; kt < 4; ++kt)
        acc = wmma_f16(a1[kt], bfrag(sV2w + hc * 128 + 32 * kt + 8 * hl), acc);
#pragma unroll
      for (int v = 0; v < 8; ++v)
        bufB[(v + 8 * hl) * 128 + hc] = (half_t)fast_tanh(acc[v]);
    }
  }

  // ---- V = 0.5 * sum(t2^2) per sample (all lanes, m = n16), vectorized ----
  float Vm = 0.f;
  {
    const v8h* pr = (const v8h*)(bufB + n16 * 128);
#pragma unroll
    for (int c = 0; c < 16; ++c) {
      v8h t = pr[c];
#pragma unroll
      for (int j = 0; j < 8; ++j) { float f = (float)t[j]; Vm += f * f; }
    }
    Vm *= 0.5f;
  }

  // ---- GEMM3: w = s @ V2_w (s = t2 - t2^3); p = w * (1 - t1^2) -> bufB (32 wmma) ----
  {
    v16h as[4];
#pragma unroll
    for (int kt = 0; kt < 4; ++kt) as[kt] = afrag_s(bufB, kt, hl, n16);
#pragma unroll
    for (int nt = 0; nt < 8; ++nt) {
      int kc = 16 * nt + n16;
      v8f acc;
#pragma unroll
      for (int v = 0; v < 8; ++v) acc[v] = 0.f;
#pragma unroll
      for (int kt = 0; kt < 4; ++kt)
        acc = wmma_f16(as[kt], bfrag(sV2T + kc * 128 + 32 * kt + 8 * hl), acc);
#pragma unroll
      for (int v = 0; v < 8; ++v) {
        int m = v + 8 * hl;
        float t1v = (float)bufA[m * 128 + kc];
        bufB[m * 128 + kc] = (half_t)(acc[v] * (1.f - t1v * t1v));
      }
    }
  }

  // ---- GEMM4: grad_V = p @ V1_w -> sGV (4 wmma) ----
  {
    v8f g;
#pragma unroll
    for (int v = 0; v < 8; ++v) g[v] = 0.f;
#pragma unroll
    for (int kt = 0; kt < 4; ++kt)
      g = wmma_f16(afrag(bufB, kt, hl, n16),
                   bfrag(sV1T + n16 * 128 + 32 * kt + 8 * hl), g);
#pragma unroll
    for (int v = 0; v < 8; ++v) sGV[(v + 8 * hl) * 16 + n16] = g[v];
  }

  // ---- K branches: K = tanh(x@Ka^T + ba) @ Kb^T + bb (2 x 12 wmma) ----
#pragma unroll
  for (int br = 0; br < 2; ++br) {
    const half_t* sWa = br ? sK2a : sK1a;
    const half_t* sWb = br ? sK2b : sK1b;
    const float* sbb = br ? sK2bb : sK1bb;
    float* dst = br ? sK2t : sK1t;

    hidden16(xa, sWa, bufB, hl, n16);
    float bb = sbb[n16];
    v8f kacc;
#pragma unroll
    for (int v = 0; v < 8; ++v) kacc[v] = bb;
#pragma unroll
    for (int kt = 0; kt < 4; ++kt)
      kacc = wmma_f16(afrag(bufB, kt, hl, n16),
                      bfrag(sWb + n16 * 128 + 32 * kt + 8 * hl), kacc);
#pragma unroll
    for (int v = 0; v < 8; ++v) dst[(v + 8 * hl) * 16 + n16] = kacc[v];
  }

  // ---- f0 = x @ A_dyn^T (1 wmma; sAd col16 == 0 so xa[8]=1 adds nothing) -> sF0 ----
  {
    v8f f;
#pragma unroll
    for (int v = 0; v < 8; ++v) f[v] = 0.f;
    v16h bf = bfrag(sAd + n16 * 32 + 8 * hl);
    f = wmma_f16(xa, bf, f);
#pragma unroll
    for (int v = 0; v < 8; ++v) sF0[(v + 8 * hl) * 16 + n16] = f[v];
  }

  // ---- per-sample finale on lanes 0..15 (m = n16, Vm already correct) ----
  if (lane < 16) {
    const int m = n16;
    const int gm = tileBase + m;
    const float* xr = x + (size_t)gm * 16;
    float u1 = sUe[0], u2 = sUe[1];
    float lg0 = 0.f, lg1 = 0.f, lf0 = 0.f;
#pragma unroll
    for (int n = 0; n < 16; ++n) {
      float dxn = xr[n] - sXg[n];
      u1 -= sK1t[m * 16 + n] * dxn;
      u2 -= sK2t[m * 16 + n] * dxn;
      float gv = sGV[m * 16 + n];
      lg0 += gv * sBd[n * 2 + 0];
      lg1 += gv * sBd[n * 2 + 1];
      lf0 += gv * sF0[m * 16 + n];
    }
    float base = lg0 * u1 + lg1 * u2 + Vm;     // CLF_LAMBDA = 1
    float r1 = lf0 + base;        r1 = r1 > 0.f ? r1 : 0.f;   // scale 1.0
    float r2 = 1.2f * lf0 + base; r2 = r2 > 0.f ? r2 : 0.f;   // scale 1.2
    float vdot = 0.5f * (r1 + r2);

    out[2 * gm]         = u1;                  // u (B,2,1)
    out[2 * gm + 1]     = u2;
    out[2 * B + 1 + gm] = Vm;                  // V (B,)
    out[3 * B + 1 + gm] = vdot;                // Vdot (B,1,1)
  }
  if (blockIdx.x == 0 && tid == 0) out[2 * B] = 0.f;   // relaxation
}

extern "C" void kernel_launch(void* const* d_in, const int* in_sizes, int n_in,
                              void* d_out, int out_size, void* d_ws, size_t ws_size,
                              hipStream_t stream) {
  const float* x      = (const float*)d_in[0];
  const float* V1_w   = (const float*)d_in[1];
  const float* V1_b   = (const float*)d_in[2];
  const float* V2_w   = (const float*)d_in[3];
  const float* V2_b   = (const float*)d_in[4];
  const float* K1a_w  = (const float*)d_in[5];
  const float* K1a_b  = (const float*)d_in[6];
  const float* K1b_w  = (const float*)d_in[7];
  const float* K1b_b  = (const float*)d_in[8];
  const float* K2a_w  = (const float*)d_in[9];
  const float* K2a_b  = (const float*)d_in[10];
  const float* K2b_w  = (const float*)d_in[11];
  const float* K2b_b  = (const float*)d_in[12];
  const float* A_dyn  = (const float*)d_in[13];
  const float* B_dyn  = (const float*)d_in[14];
  const float* x_goal = (const float*)d_in[15];
  const float* u_eq   = (const float*)d_in[16];
  float* out = (float*)d_out;

  const int B = in_sizes[0] / 16;
  // LDS: f16 weights (padded + transposed copies) + per-wave staging + f32 tables
  size_t smem = (size_t)(2 * 128 * 128 + 3 * 128 * 32 + 3 * 16 * 128 + 16 * 32) * 2
              + (size_t)WAVES * 6144 * 2
              + (size_t)(128 + 2 * 16 + 32 + 16 + 2) * 4;
  dim3 grid((B + SPB - 1) / SPB);
  clf_k_qp_kernel<<<grid, BDIM, smem, stream>>>(
      x, V1_w, V1_b, V2_w, V2_b, K1a_w, K1a_b, K1b_w, K1b_b,
      K2a_w, K2a_b, K2b_w, K2b_b, A_dyn, B_dyn, x_goal, u_eq, out, B);
}
